// RNNPolicy_45509473468531
// MI455X (gfx1250) — matrix operand order repaired and checked by reading, not compiled, not run on hardware
//
#include <hip/hip_runtime.h>

#define SEQ   1024
#define BATCH 256
#define OBSZ  64
#define HID   256
#define ACTZ  32

typedef __attribute__((ext_vector_type(16))) __bf16 v16bf;
typedef __attribute__((ext_vector_type(8)))  float  v8f;

__device__ __forceinline__ v8f wmma_bf16(v16bf a, v16bf b, v8f c) {
  // D = A(16x32 bf16) * B(32x16 bf16) + C(16x16 f32)
  return __builtin_amdgcn_wmma_f32_16x16x32_bf16(false, a, false, b, (short)0, c,
                                                 false, false);
}

__device__ __forceinline__ v8f splat8(float x) {
  v8f v;
#pragma unroll
  for (int i = 0; i < 8; ++i) v[i] = x;
  return v;
}

// ---------------------------------------------------------------------------
// Pack phase: fp32 -> bf16, rearranged into exact WMMA operand layouts so the
// persistent kernel's hot loop only issues contiguous 32B-per-lane loads.
// B layout (ISA 7.12.2, 16-bit B KxN=32x16): lane = N + 16*((K>>3)&1), in-lane K
//   order same as A: K = (e&7) + 16*(e>>3) + 8*(lane>>4).
// A layout (16-bit A MxK=16x32): lane<16 holds M=lane, K in {0..7,16..23};
//   lane>=16 holds M=lane-16, K in {8..15,24..31}; in-lane element e maps to
//   K = (e&7) + 16*(e>>3) + 8*(lane>>4).
// ---------------------------------------------------------------------------

// W is [N, K] row-major; computing x @ W.T means B[k][n] = W[n][k].
__global__ void pack_weight(const float* __restrict__ W, __bf16* __restrict__ dst,
                            int ntiles, int ktiles, int K) {
  int chunk = blockIdx.x * blockDim.x + threadIdx.x;     // [nt][kt][lane]
  int total = ntiles * ktiles * 32;
  if (chunk >= total) return;
  int lane = chunk & 31;
  int kt   = (chunk >> 5) % ktiles;
  int nt   = (chunk >> 5) / ktiles;
  int n    = nt * 16 + (lane & 15);
  int kb   = kt * 32 + (lane >> 4) * 8;
  v16bf v;
#pragma unroll
  for (int e = 0; e < 16; ++e) v[e] = (__bf16)W[n * K + kb + (e & 7) + 16 * (e >> 3)];
  *(v16bf*)(dst + chunk * 16) = v;
}

// obs [SEQ, BATCH, OBSZ] -> A-operand tiles [t][btile(16)][kt(2)][lane(32)][16]
__global__ void pack_obs(const float* __restrict__ obs, __bf16* __restrict__ dst) {
  int chunk = blockIdx.x * blockDim.x + threadIdx.x;     // 2^20 chunks
  if (chunk >= SEQ * 16 * 2 * 32) return;
  int lane = chunk & 31;
  int kt   = (chunk >> 5) & 1;
  int bt   = (chunk >> 6) & 15;
  int t    = chunk >> 10;
  int b    = bt * 16 + (lane & 15);
  int kb   = kt * 32 + 8 * (lane >> 4);
  const float* src = obs + ((size_t)t * BATCH + b) * OBSZ;
  v16bf v;
#pragma unroll
  for (int e = 0; e < 16; ++e) v[e] = (__bf16)src[kb + (e & 7) + 16 * (e >> 3)];
  *(v16bf*)(dst + (size_t)chunk * 16) = v;
}

// ---------------------------------------------------------------------------
// Persistent RNN kernel: 16 WGs (one 16-row batch tile each), 8 wave32s per WG
// (each wave owns 32 hidden columns = two 16x16 N-tiles). Weights pinned in
// VGPRs (W_ih0/W_ih1/W_hh1/W_out slices) and LDS (W_hh0, 128 KB). h0/h1 are
// double-buffered bf16 LDS tiles stored directly in A-operand layout.
// Two workgroup barriers per timestep.
// ---------------------------------------------------------------------------
__global__ __launch_bounds__(256, 1) void rnn_persistent(
    const __bf16* __restrict__ obs_p,
    const __bf16* __restrict__ wih0_p, const __bf16* __restrict__ whh0_p,
    const __bf16* __restrict__ wih1_p, const __bf16* __restrict__ whh1_p,
    const __bf16* __restrict__ wout_p,
    const float* __restrict__ b_ih0, const float* __restrict__ b_ih1,
    const float* __restrict__ b_out, float* __restrict__ out) {
  __shared__ __align__(32) __bf16 lds_whh0[16 * 8 * 32 * 16];  // 128 KB (CDNA5 320KB LDS)
  __shared__ __align__(32) __bf16 lds_h0[2][8 * 32 * 16];      // 16 KB double-buffered
  __shared__ __align__(32) __bf16 lds_h1[2][8 * 32 * 16];      // 16 KB

  const int tid  = threadIdx.x;
  const int w    = tid >> 5;        // wave 0..7 (wave32)
  const int lane = tid & 31;
  const int blk  = blockIdx.x;      // batch tile
  const int dlo  = lane & 15, dhi = lane >> 4;

  // W_hh0 -> LDS (already in per-slice B-operand order)
  {
    const uint4* s = (const uint4*)whh0_p;
    uint4* d = (uint4*)lds_whh0;
#pragma unroll
    for (int i = 0; i < 32; ++i) d[tid + 256 * i] = s[tid + 256 * i];
  }
  // zero-init hidden state buffers
  {
    unsigned* z0 = (unsigned*)lds_h0;
    unsigned* z1 = (unsigned*)lds_h1;
#pragma unroll
    for (int i = 0; i < 16; ++i) { z0[tid + 256 * i] = 0u; z1[tid + 256 * i] = 0u; }
  }

  // register-resident weight slices (live across the whole scan)
  v16bf wih0[2][2], wih1[2][8], whh1[2][8], wout[8];
#pragma unroll
  for (int j = 0; j < 2; ++j)
#pragma unroll
    for (int kt = 0; kt < 2; ++kt)
      wih0[j][kt] = *(const v16bf*)(wih0_p + (((2 * w + j) * 2 + kt) * 32 + lane) * 16);
#pragma unroll
  for (int j = 0; j < 2; ++j)
#pragma unroll
    for (int kt = 0; kt < 8; ++kt) {
      wih1[j][kt] = *(const v16bf*)(wih1_p + (((2 * w + j) * 8 + kt) * 32 + lane) * 16);
      whh1[j][kt] = *(const v16bf*)(whh1_p + (((2 * w + j) * 8 + kt) * 32 + lane) * 16);
    }
#pragma unroll
  for (int kt = 0; kt < 8; ++kt)
    wout[kt] = *(const v16bf*)(wout_p + (((w & 1) * 8 + kt) * 32 + lane) * 16);

  const int n0 = (2 * w) * 16 + dlo, n1 = n0 + 16;
  const float bih0_0 = b_ih0[n0], bih0_1 = b_ih0[n1];
  const float bih1_0 = b_ih1[n0], bih1_1 = b_ih1[n1];
  const float bo = b_out[(w & 1) * 16 + dlo];

  __syncthreads();

  // scatter indices: D(m,n) of tile j -> A-layout slot (lane_a = r+lbase, e = ebase+8j)
  const int lbase = 8 * dhi + 16 * ((dlo >> 3) & 1);
  const int ebase = dlo & 7;

  for (int t = 0; t < SEQ; ++t) {
    const int rb = 1 - (t & 1), wb = t & 1;
    if (t + 1 < SEQ) {  // gfx1250 global_prefetch_b8 for next step's obs tile
      __builtin_prefetch(obs_p + ((((t + 1) * 16 + blk) * 2 + 0) * 32 + lane) * 16, 0, 0);
      __builtin_prefetch(obs_p + ((((t + 1) * 16 + blk) * 2 + 1) * 32 + lane) * 16, 0, 0);
    }

    // ---- layer 0: h0 = relu(x W_ih0^T + b + h0_prev W_hh0^T) ----
    v8f a0 = splat8(bih0_0), a1 = splat8(bih0_1);
#pragma unroll
    for (int kt = 0; kt < 2; ++kt) {
      v16bf a = *(const v16bf*)(obs_p + (((t * 16 + blk) * 2 + kt) * 32 + lane) * 16);
      a0 = wmma_bf16(a, wih0[0][kt], a0);
      a1 = wmma_bf16(a, wih0[1][kt], a1);
    }
#pragma unroll
    for (int kt = 0; kt < 8; ++kt) {
      v16bf a  = *(const v16bf*)&lds_h0[rb][(kt * 32 + lane) * 16];
      v16bf bA = *(const v16bf*)&lds_whh0[(((2 * w) * 8 + kt) * 32 + lane) * 16];
      v16bf bB = *(const v16bf*)&lds_whh0[(((2 * w + 1) * 8 + kt) * 32 + lane) * 16];
      a0 = wmma_bf16(a, bA, a0);
      a1 = wmma_bf16(a, bB, a1);
    }
    {
      __bf16* hb = &lds_h0[wb][w * 32 * 16];
#pragma unroll
      for (int r = 0; r < 8; ++r) {
        __bf16* p = hb + (r + lbase) * 16 + ebase;
        p[0] = (__bf16)fmaxf(a0[r], 0.0f);
        p[8] = (__bf16)fmaxf(a1[r], 0.0f);
      }
    }
    __syncthreads();

    // ---- layer 1: h1 = relu(h0 W_ih1^T + b + h1_prev W_hh1^T) ----
    v8f c0 = splat8(bih1_0), c1 = splat8(bih1_1);
#pragma unroll
    for (int kt = 0; kt < 8; ++kt) {
      v16bf a = *(const v16bf*)&lds_h0[wb][(kt * 32 + lane) * 16];
      c0 = wmma_bf16(a, wih1[0][kt], c0);
      c1 = wmma_bf16(a, wih1[1][kt], c1);
    }
#pragma unroll
    for (int kt = 0; kt < 8; ++kt) {
      v16bf a = *(const v16bf*)&lds_h1[rb][(kt * 32 + lane) * 16];
      c0 = wmma_bf16(a, whh1[0][kt], c0);
      c1 = wmma_bf16(a, whh1[1][kt], c1);
    }
    {
      __bf16* hb = &lds_h1[wb][w * 32 * 16];
#pragma unroll
      for (int r = 0; r < 8; ++r) {
        __bf16* p = hb + (r + lbase) * 16 + ebase;
        p[0] = (__bf16)fmaxf(c0[r], 0.0f);
        p[8] = (__bf16)fmaxf(c1[r], 0.0f);
      }
    }
    __syncthreads();

    // ---- head: sigmoid(h1 W_out^T + b). WMMA runs full-EXEC on all waves
    // (EXEC must be all-1s for WMMA); only waves 0,1 own real N-tiles & store.
    v8f o = splat8(bo);
#pragma unroll
    for (int kt = 0; kt < 8; ++kt) {
      v16bf a = *(const v16bf*)&lds_h1[wb][(kt * 32 + lane) * 16];
      o = wmma_bf16(a, wout[kt], o);
    }
    if (w < 2) {
      const int n = w * 16 + dlo;
      float* orow = out + (size_t)t * (BATCH * ACTZ) + (blk * 16 + 8 * dhi) * ACTZ + n;
#pragma unroll
      for (int r = 0; r < 8; ++r)
        orow[r * ACTZ] = 1.0f / (1.0f + __expf(-o[r]));
    }
  }

  // final hidden states (fp32, inverse of A-operand layout; last write buf = 1)
  {
    float* h0o = out + (size_t)SEQ * BATCH * ACTZ;
    float* h1o = h0o + BATCH * HID;
    const int kt = tid >> 5, la = tid & 31;
    const int m = la & 15, b = blk * 16 + m, khi = 8 * (la >> 4);
#pragma unroll
    for (int e = 0; e < 16; ++e) {
      int k = kt * 32 + (e & 7) + 16 * (e >> 3) + khi;
      h0o[(size_t)b * HID + k] = (float)lds_h0[1][(kt * 32 + la) * 16 + e];
      h1o[(size_t)b * HID + k] = (float)lds_h1[1][(kt * 32 + la) * 16 + e];
    }
  }
}

extern "C" void kernel_launch(void* const* d_in, const int* in_sizes, int n_in,
                              void* d_out, int out_size, void* d_ws, size_t ws_size,
                              hipStream_t stream) {
  (void)in_sizes; (void)n_in; (void)out_size; (void)ws_size;
  const float* obs   = (const float*)d_in[0];
  const float* W_ih0 = (const float*)d_in[1];
  const float* b_ih0 = (const float*)d_in[2];
  const float* W_hh0 = (const float*)d_in[3];
  const float* W_ih1 = (const float*)d_in[4];
  const float* b_ih1 = (const float*)d_in[5];
  const float* W_hh1 = (const float*)d_in[6];
  const float* W_out = (const float*)d_in[7];
  const float* b_out = (const float*)d_in[8];

  // workspace layout (bf16 elements): ~34 MB total
  __bf16* ws     = (__bf16*)d_ws;
  __bf16* obs_p  = ws;                  // 1024*16*2*32*16 = 16,777,216
  __bf16* wih0_p = ws + 16777216;       // 16*2*32*16  = 16,384
  __bf16* whh0_p = wih0_p + 16384;      // 16*8*32*16  = 65,536
  __bf16* wih1_p = whh0_p + 65536;      // 65,536
  __bf16* whh1_p = wih1_p + 65536;      // 65,536
  __bf16* wout_p = whh1_p + 65536;      // 2*8*32*16   = 8,192

  pack_obs<<<4096, 256, 0, stream>>>(obs, obs_p);
  pack_weight<<<4, 256, 0, stream>>>(W_ih0, wih0_p, 16, 2, OBSZ);
  pack_weight<<<16, 256, 0, stream>>>(W_hh0, whh0_p, 16, 8, HID);
  pack_weight<<<16, 256, 0, stream>>>(W_ih1, wih1_p, 16, 8, HID);
  pack_weight<<<16, 256, 0, stream>>>(W_hh1, whh1_p, 16, 8, HID);
  pack_weight<<<2, 256, 0, stream>>>(W_out, wout_p, 2, 8, HID);

  rnn_persistent<<<16, 256, 0, stream>>>(obs_p, wih0_p, whh0_p, wih1_p, whh1_p,
                                         wout_p, b_ih0, b_ih1, b_out, (float*)d_out);
}
